// CausalSelfAttention_4054449127966
// MI455X (gfx1250) — matrix operand order repaired and checked by reading, not compiled, not run on hardware
//
#include <hip/hip_runtime.h>
#include <hip/hip_bf16.h>
#include <math.h>

typedef __attribute__((ext_vector_type(16))) _Float16 v16h;
typedef __attribute__((ext_vector_type(8)))  _Float16 v8h;
typedef __attribute__((ext_vector_type(8)))  float    v8f;

#define NEG_INF (-1.0e30f)

__device__ __forceinline__ v16h cat8(v8h lo, v8h hi) {
  return __builtin_shufflevector(lo, hi, 0,1,2,3,4,5,6,7,8,9,10,11,12,13,14,15);
}
__device__ __forceinline__ v8f zero8() {
  v8f z;
#pragma unroll
  for (int i = 0; i < 8; ++i) z[i] = 0.0f;
  return z;
}
__device__ __forceinline__ v8f wmma16(v16h a, v16h b, v8f c) {
  // D = A(16x32 f16) x B(32x16 f16) + C(16x16 f32)
  return __builtin_amdgcn_wmma_f32_16x16x32_f16(false, a, false, b, (short)0, c, false, false);
}

// CDNA5 async global->LDS copy (16B per lane), tracked with ASYNCcnt.
// Generic LDS pointers carry the LDS byte offset in their low 32 bits (ISA 10.2).
__device__ __forceinline__ void async_copy_b128(void* lds, const void* g) {
  unsigned l = (unsigned)(unsigned long long)(uintptr_t)lds;
  unsigned long long a = (unsigned long long)(uintptr_t)g;
  asm volatile("global_load_async_to_lds_b128 %0, %1, off"
               :: "v"(l), "v"(a) : "memory");
}
__device__ __forceinline__ void wait_async0() {
  asm volatile("s_wait_asynccnt 0" ::: "memory");
}

// ---------------------------------------------------------------------------
// Kernel 0a: HoPE relative-position bias table bias_rel[8][1024].
// active = min(244,128)=128 pairs -> heads 0..3 get 32 freq pairs each,
// heads 4..7 get only the position-independent constant.
// ---------------------------------------------------------------------------
__global__ void bias_kernel(const float* __restrict__ pos_ind,
                            float* __restrict__ bias_rel) {
  const int h = blockIdx.x;      // 0..7
  const int tid = threadIdx.x;   // 256 threads
  if (h < 4) {
    for (int r = tid; r < 1024; r += 256) {
      float acc = 0.0f;
      #pragma unroll 4
      for (int j = 0; j < 32; ++j) {
        int p = h * 32 + j;
        // freq = 10000^(-p/256) = exp(-p * ln(10000)/256)
        float f = __expf(-(float)p * 0.035977892f);
        float sn, cs;
        __sincosf((float)r * f, &sn, &cs);
        acc += sn + cs;
      }
      bias_rel[h * 1024 + r] = acc;
    }
  } else {
    float c = 0.0f;
    #pragma unroll 4
    for (int j = 0; j < 32; ++j) {
      int i = (h - 4) * 32 + j;              // pi_sum index
      c += pos_ind[i * 2 + 0] + pos_ind[i * 2 + 1];
    }
    for (int r = tid; r < 1024; r += 256) bias_rel[h * 1024 + r] = c;
  }
}

// ---------------------------------------------------------------------------
// Kernel 0b: x (fp32 [4096,512]) -> fp16 copy, once.
// ---------------------------------------------------------------------------
__global__ __launch_bounds__(256) void cvt_x_kernel(const float* __restrict__ x,
                                                    _Float16* __restrict__ xh) {
  size_t i = ((size_t)blockIdx.x * 256 + threadIdx.x) * 8;
  float4 a = *(const float4*)(x + i);
  float4 b = *(const float4*)(x + i + 4);
  v8h h;
  h[0] = (_Float16)a.x; h[1] = (_Float16)a.y; h[2] = (_Float16)a.z; h[3] = (_Float16)a.w;
  h[4] = (_Float16)b.x; h[5] = (_Float16)b.y; h[6] = (_Float16)b.z; h[7] = (_Float16)b.w;
  *(v8h*)(xh + i) = h;
}

// ---------------------------------------------------------------------------
// Kernel 0c: weights fp32 [512,512] (k,n) -> fp16 transposed [n][k], once.
// blockIdx.y selects Wq/Wk/Wv/Wp.
// ---------------------------------------------------------------------------
__global__ __launch_bounds__(256) void cvt_wT_kernel(
    const float* __restrict__ Wq, const float* __restrict__ Wk,
    const float* __restrict__ Wv, const float* __restrict__ Wp,
    _Float16* __restrict__ WtAll) {
  const int wsel = blockIdx.y;
  const float* W = (wsel == 0) ? Wq : (wsel == 1) ? Wk : (wsel == 2) ? Wv : Wp;
  _Float16* Wt = WtAll + (size_t)wsel * 512 * 512;
  int idx = blockIdx.x * 256 + threadIdx.x;   // 0..32767
  int n  = idx >> 6;                          // 0..511
  int k8 = (idx & 63) * 8;
  v8h h;
  #pragma unroll
  for (int j = 0; j < 8; ++j)
    h[j] = (_Float16)W[(size_t)(k8 + j) * 512 + n];
  *(v8h*)(&Wt[(size_t)n * 512 + k8]) = h;
}

// ---------------------------------------------------------------------------
// Unified fp16 GEMM: C(128x64 block) = A[4096,512] x Wt^T + bias.
// 8 waves, each a 32x32 sub-tile (4 WMMAs / 32-wide K-step).
// Tiles staged to LDS with global_load_async_to_lds_b128.
// mode 0: q  (fp16 *0.125 -> [B*H,1024,64])
// mode 1: k  (fp16        -> [B*H,1024,64])
// mode 2: v  (gated: fp32 -> d_out v section; fp16 -> vT [B*H,64,1024])
// mode 3: out projection  (fp32 + bias -> d_out)
// ---------------------------------------------------------------------------
__global__ __launch_bounds__(256) void gemm16_kernel(
    const _Float16* __restrict__ A,     // [4096,512] fp16
    const _Float16* __restrict__ Wt,    // [512,512] fp16, layout [n][k]
    const float* __restrict__ bias,     // [512]
    int mode,
    _Float16* __restrict__ qk_out,      // modes 0/1
    const float* __restrict__ v1,       // mode 2
    const float* __restrict__ lamb1p,
    const float* __restrict__ lamb2p,
    float* __restrict__ v_out,          // mode 2
    _Float16* __restrict__ vT_out,      // mode 2
    float* __restrict__ f_out)          // mode 3
{
  __shared__ __align__(16) _Float16 xs[128 * 32];  // A tile [row][k]
  __shared__ __align__(16) _Float16 wsm[64 * 32];  // W tile [n][k]

  const int tid  = threadIdx.x;
  const int lane = tid & 31;
  const int wave = tid >> 5;      // 0..7
  const int wrow = wave >> 1;     // 0..3 -> rows wrow*32
  const int wcol = wave & 1;      // 0..1 -> cols wcol*32
  const int m0 = blockIdx.x * 128;
  const int n0 = blockIdx.y * 64;
  const int am = lane & 15;
  const int ah = lane >> 4;

  v8f acc[2][2];
  #pragma unroll
  for (int mt = 0; mt < 2; ++mt)
    #pragma unroll
    for (int nt = 0; nt < 2; ++nt) acc[mt][nt] = zero8();

  const int ar  = tid >> 1;            // 0..127 : A tile row
  const int ac8 = (tid & 1) * 16;      // 0 or 16
  const int bn  = tid & 63;            // W tile n
  const int bk8 = (tid >> 6) * 8;      // 0,8,16,24

  for (int kk = 0; kk < 512; kk += 32) {
    // ---- async-stage A (128x32) and W (64x32) tiles ----
    async_copy_b128(&xs[ar * 32 + ac8],     A  + (size_t)(m0 + ar) * 512 + kk + ac8);
    async_copy_b128(&xs[ar * 32 + ac8 + 8], A  + (size_t)(m0 + ar) * 512 + kk + ac8 + 8);
    async_copy_b128(&wsm[bn * 32 + bk8],    Wt + (size_t)(n0 + bn) * 512 + kk + bk8);
    wait_async0();
    __syncthreads();

    // ---- fragments ----
    v16h af[2], bf[2];
    #pragma unroll
    for (int mt = 0; mt < 2; ++mt) {
      const _Float16* p = &xs[(wrow * 32 + mt * 16 + am) * 32 + ah * 8];
      af[mt] = cat8(*(const v8h*)(p), *(const v8h*)(p + 16));
    }
    #pragma unroll
    for (int nt = 0; nt < 2; ++nt) {
      const _Float16* p = &wsm[(wcol * 32 + nt * 16 + am) * 32 + ah * 8];
      bf[nt] = cat8(*(const v8h*)(p), *(const v8h*)(p + 16));
    }
    #pragma unroll
    for (int mt = 0; mt < 2; ++mt)
      #pragma unroll
      for (int nt = 0; nt < 2; ++nt)
        acc[mt][nt] = wmma16(af[mt], bf[nt], acc[mt][nt]);
    __syncthreads();
  }

  // ---- epilogue ----
  const float l1 = (mode == 2) ? lamb1p[0] : 0.0f;
  const float l2 = (mode == 2) ? lamb2p[0] : 0.0f;
  #pragma unroll
  for (int mt = 0; mt < 2; ++mt) {
    #pragma unroll
    for (int r = 0; r < 8; ++r) {
      int mg = m0 + wrow * 32 + mt * 16 + r + 8 * ah;  // global row = b*1024+t
      int bb = mg >> 10, tt = mg & 1023;
      #pragma unroll
      for (int nt = 0; nt < 2; ++nt) {
        int ng = n0 + wcol * 32 + nt * 16 + am;        // 0..511
        float val = acc[mt][nt][r] + bias[ng];
        int hh = ng >> 6, hd = ng & 63;
        if (mode == 0) {
          qk_out[(((size_t)(bb * 8 + hh)) * 1024 + tt) * 64 + hd] =
              (_Float16)(val * 0.125f);                // fold 1/sqrt(64)
        } else if (mode == 1) {
          qk_out[(((size_t)(bb * 8 + hh)) * 1024 + tt) * 64 + hd] = (_Float16)val;
        } else if (mode == 2) {
          float vv = l1 * val + l2 * v1[(size_t)mg * 512 + ng];
          v_out[(size_t)mg * 512 + ng] = vv;
          vT_out[(((size_t)(bb * 8 + hh)) * 64 + hd) * 1024 + tt] = (_Float16)vv;
        } else {
          f_out[(size_t)mg * 512 + ng] = val;
        }
      }
    }
  }
}

// ---------------------------------------------------------------------------
// Kernel 2: flash attention per (b,h). Block = 128 thr (4 waves), 64 q rows.
// Wave w owns rows t0+16w..+15.  s-blocks of 64, causal, HoPE bias via table.
// ---------------------------------------------------------------------------
__global__ __launch_bounds__(128) void attn_kernel(
    const _Float16* __restrict__ q,    // [B*H,1024,64] (pre-scaled)
    const _Float16* __restrict__ k,    // [B*H,1024,64]
    const _Float16* __restrict__ vT,   // [B*H,64,1024]
    const float* __restrict__ bias_rel,// [8,1024]
    _Float16* __restrict__ attO)       // [4096,512]
{
  __shared__ __align__(16) float    bias_s[1024];
  __shared__ __align__(16) _Float16 pbuf[4][16 * 64];

  const int tid  = threadIdx.x;
  const int lane = tid & 31;
  const int wave = tid >> 5;          // 0..3
  const int t0 = blockIdx.x * 64;
  const int bh = blockIdx.y;          // b*8 + h
  const int h  = bh & 7;
  const int b_ = bh >> 3;

  for (int i = tid; i < 1024; i += 128) bias_s[i] = bias_rel[h * 1024 + i];
  __syncthreads();

  const _Float16* qp = q  + (size_t)bh * 1024 * 64;
  const _Float16* kp = k  + (size_t)bh * 1024 * 64;
  const _Float16* vp = vT + (size_t)bh * 64 * 1024;

  const int am = lane & 15;
  const int ah = lane >> 4;

  // Q fragments for the whole kernel (rows t0+16*wave+am, d in 2 halves)
  v16h aq[2];
  {
    int trow = t0 + wave * 16 + am;
    #pragma unroll
    for (int ks = 0; ks < 2; ++ks) {
      const _Float16* qr = qp + (size_t)trow * 64 + ks * 32 + ah * 8;
      v8h lo = *(const v8h*)(qr);
      v8h hi = *(const v8h*)(qr + 16);
      aq[ks] = cat8(lo, hi);
    }
  }

  v8f o[4];
  #pragma unroll
  for (int nt = 0; nt < 4; ++nt) o[nt] = zero8();
  float mstat[8], lstat[8];
  #pragma unroll
  for (int r = 0; r < 8; ++r) { mstat[r] = NEG_INF; lstat[r] = 0.0f; }

  const int trow_c = t0 + wave * 16 + 8 * ah;   // + r gives this lane's C-row
  const int nsb = blockIdx.x + 1;

  for (int sb = 0; sb < nsb; ++sb) {
    const int s0 = sb * 64;

    // ---- S = Q K^T  (16 x 64) : 8 WMMAs ----
    v8f sa[4];
    #pragma unroll
    for (int nt = 0; nt < 4; ++nt) sa[nt] = zero8();
    #pragma unroll
    for (int ks = 0; ks < 2; ++ks) {
      #pragma unroll
      for (int nt = 0; nt < 4; ++nt) {
        int srow = s0 + nt * 16 + am;           // B-fragment column -> K row s
        const _Float16* kr = kp + (size_t)srow * 64 + ks * 32 + ah * 8;
        v8h lo = *(const v8h*)(kr);
        v8h hi = *(const v8h*)(kr + 16);
        sa[nt] = wmma16(aq[ks], cat8(lo, hi), sa[nt]);
      }
    }

    // ---- bias + causal mask + online softmax stats ----
    float alpha[8];
    #pragma unroll
    for (int r = 0; r < 8; ++r) {
      int tg = trow_c + r;
      float mx = NEG_INF;
      #pragma unroll
      for (int nt = 0; nt < 4; ++nt) {
        int sg = s0 + nt * 16 + am;
        int rel = tg - sg;
        float val = (rel >= 0) ? (sa[nt][r] + bias_s[rel & 1023]) : NEG_INF;
        sa[nt][r] = val;
        mx = fmaxf(mx, val);
      }
      #pragma unroll
      for (int off = 1; off < 16; off <<= 1)
        mx = fmaxf(mx, __shfl_xor(mx, off, 32));
      float newm = fmaxf(mstat[r], mx);
      float al = __expf(mstat[r] - newm);
      alpha[r] = al;
      float rsum = 0.0f;
      #pragma unroll
      for (int nt = 0; nt < 4; ++nt) {
        float p = __expf(sa[nt][r] - newm);
        sa[nt][r] = p;
        rsum += p;
      }
      #pragma unroll
      for (int off = 1; off < 16; off <<= 1)
        rsum += __shfl_xor(rsum, off, 32);
      lstat[r] = lstat[r] * al + rsum;
      mstat[r] = newm;
    }

    // ---- P -> per-wave LDS (fp16) for A-fragment transpose ----
    _Float16* pb = &pbuf[wave][0];
    #pragma unroll
    for (int nt = 0; nt < 4; ++nt)
      #pragma unroll
      for (int r = 0; r < 8; ++r)
        pb[(r + 8 * ah) * 64 + nt * 16 + am] = (_Float16)sa[nt][r];

    // ---- rescale O by alpha (per-lane rows match C layout) ----
    #pragma unroll
    for (int nt = 0; nt < 4; ++nt)
      #pragma unroll
      for (int r = 0; r < 8; ++r)
        o[nt][r] *= alpha[r];

    // ---- O += P V : 8 WMMAs ----
    #pragma unroll
    for (int ks = 0; ks < 2; ++ks) {
      v8h plo = *(v8h*)(&pbuf[wave][am * 64 + ks * 32 + ah * 8]);
      v8h phi = *(v8h*)(&pbuf[wave][am * 64 + ks * 32 + 16 + ah * 8]);
      v16h ap = cat8(plo, phi);
      #pragma unroll
      for (int nt = 0; nt < 4; ++nt) {
        const _Float16* vr = vp + (size_t)(nt * 16 + am) * 1024 + s0 + ks * 32 + ah * 8;
        v8h lo = *(const v8h*)(vr);
        v8h hi = *(const v8h*)(vr + 16);
        o[nt] = wmma16(ap, cat8(lo, hi), o[nt]);
      }
    }
  }

  // ---- normalize and store attO (fp16, [b*1024+t][h*64+d]) ----
  #pragma unroll
  for (int r = 0; r < 8; ++r) {
    float inv = 1.0f / lstat[r];
    int tg = trow_c + r;
    size_t row = (size_t)b_ * 1024 + tg;
    #pragma unroll
    for (int nt = 0; nt < 4; ++nt)
      attO[row * 512 + h * 64 + nt * 16 + am] = (_Float16)(o[nt][r] * inv);
  }
}

// ---------------------------------------------------------------------------
extern "C" void kernel_launch(void* const* d_in, const int* in_sizes, int n_in,
                              void* d_out, int out_size, void* d_ws, size_t ws_size,
                              hipStream_t stream) {
  (void)in_sizes; (void)n_in; (void)out_size; (void)ws_size;

  const float* x       = (const float*)d_in[0];
  const float* v1      = (const float*)d_in[1];
  const float* Wq      = (const float*)d_in[2];
  const float* bq      = (const float*)d_in[3];
  const float* Wk      = (const float*)d_in[4];
  const float* bk      = (const float*)d_in[5];
  const float* Wv      = (const float*)d_in[6];
  const float* bv      = (const float*)d_in[7];
  const float* Wp      = (const float*)d_in[8];
  const float* bp      = (const float*)d_in[9];
  const float* lamb1   = (const float*)d_in[10];
  const float* lamb2   = (const float*)d_in[11];
  const float* pos_ind = (const float*)d_in[12];

  float* out  = (float*)d_out;                       // [4096*512]
  float* vout = out + (size_t)4096 * 512;            // [4096*512]

  char* w = (char*)d_ws;
  float*    bias_rel = (float*)w;                    // 32 KB
  _Float16* xh  = (_Float16*)(w + 32768);            // 4 MB  fp16 x
  _Float16* WtA = xh + (size_t)2097152;              // 2 MB  4 transposed weights
  _Float16* qf  = WtA + (size_t)4 * 262144;          // 4 MB each
  _Float16* kf  = qf + (size_t)2097152;
  _Float16* vT  = kf + (size_t)2097152;
  _Float16* aO  = vT + (size_t)2097152;

  _Float16* WtQ = WtA + (size_t)0 * 262144;
  _Float16* WtK = WtA + (size_t)1 * 262144;
  _Float16* WtV = WtA + (size_t)2 * 262144;
  _Float16* WtP = WtA + (size_t)3 * 262144;

  bias_kernel<<<8, 256, 0, stream>>>(pos_ind, bias_rel);
  cvt_x_kernel<<<1024, 256, 0, stream>>>(x, xh);
  cvt_wT_kernel<<<dim3(128, 4), 256, 0, stream>>>(Wq, Wk, Wv, Wp, WtA);

  dim3 gg(32, 8);  // M/128 x N/64
  gemm16_kernel<<<gg, 256, 0, stream>>>(xh, WtQ, bq, 0, qf, v1, lamb1, lamb2, vout, vT, out);
  gemm16_kernel<<<gg, 256, 0, stream>>>(xh, WtK, bk, 1, kf, v1, lamb1, lamb2, vout, vT, out);
  gemm16_kernel<<<gg, 256, 0, stream>>>(xh, WtV, bv, 2, qf, v1, lamb1, lamb2, vout, vT, out);

  attn_kernel<<<dim3(16, 32), 128, 0, stream>>>(qf, kf, vT, bias_rel, aO);

  gemm16_kernel<<<gg, 256, 0, stream>>>(aO, WtP, bp, 3, qf, v1, lamb1, lamb2, vout, vT, out);
}
